// DPQ_19232863551821
// MI455X (gfx1250) — compile-verified, matrix operand order checked by reading, not run on hardware
//
#include <hip/hip_runtime.h>

typedef __attribute__((ext_vector_type(16))) __bf16 v16bf;
typedef __attribute__((ext_vector_type(8)))  __bf16 v8bf;
typedef __attribute__((ext_vector_type(8)))  float  v8f;

#define N_TOTAL 65536
#define D_TOTAL 512
#define M_SUB   8
#define K_CODES 256
#define DSUB    64   // D_TOTAL / M_SUB

// ---------------------------------------------------------------------------
// Kernel 0: codebook f32 -> bf16 (into ws) + c2[m][k] = sum_d c^2 (into ws)
// ---------------------------------------------------------------------------
__global__ __launch_bounds__(256) void pq_prep_kernel(
    const float* __restrict__ codebook, __bf16* __restrict__ cbw,
    float* __restrict__ c2w)
{
    int tid = blockIdx.x * 256 + threadIdx.x;      // one (m,k) codeword
    if (tid >= M_SUB * K_CODES) return;
    const float* src = codebook + (size_t)tid * DSUB;
    __bf16* dst = cbw + (size_t)tid * DSUB;
    float s = 0.0f;
#pragma unroll
    for (int d = 0; d < DSUB; ++d) {
        float v = src[d];
        s = __builtin_fmaf(v, v, s);
        dst[d] = (__bf16)v;
    }
    c2w[tid] = s;
}

// ---------------------------------------------------------------------------
// Kernel 1: WMMA distance + argmin -> codes
// Block = 256 threads (8 waves), handles 16 rows of x. Wave w owns subspace m=w.
// C-tile mapping (f32 16x16): element j of acc, lane l -> row (j + 8*(l>>4)),
// code column (ktile*16 + (l&15)).  A 16x32 bf16 per-lane data = two contiguous
// 8-element (16B) chunks of one x row; B likewise of one codeword row.
// ---------------------------------------------------------------------------
__global__ __launch_bounds__(256) void pq_codes_kernel(
    const float* __restrict__ x, const __bf16* __restrict__ cbw,
    const float* __restrict__ c2w, int* __restrict__ codes_out)
{
    __shared__ __align__(16) __bf16 xs[16 * D_TOTAL];   // 16 KB

    const int t  = threadIdx.x;
    const int n0 = blockIdx.x * 16;

    // cooperative load + f32->bf16 convert of the 16-row x tile
#pragma unroll
    for (int i = 0; i < 32; ++i) {
        int idx = t + i * 256;                       // 0 .. 8191, coalesced
        xs[idx] = (__bf16)x[(size_t)n0 * D_TOTAL + idx];
    }
    __syncthreads();

    const int m    = t >> 5;        // wave id == subspace
    const int lane = t & 31;
    const int h    = lane >> 4;     // lane half (0/1)
    const int c    = lane & 15;     // column within tile

    // A fragments: row = c, K-dim = d within subspace m; dstep in {0,1}
    v16bf afrag[2];
#pragma unroll
    for (int dstep = 0; dstep < 2; ++dstep) {
        const __bf16* ab = &xs[c * D_TOTAL + m * DSUB + dstep * 32 + h * 8];
        v8bf lo = *(const v8bf*)(ab);
        v8bf hi = *(const v8bf*)(ab + 16);
#pragma unroll
        for (int i = 0; i < 8; ++i) { afrag[dstep][i] = lo[i]; afrag[dstep][i + 8] = hi[i]; }
    }

    float minv[8];
    int   mini[8];
#pragma unroll
    for (int j = 0; j < 8; ++j) { minv[j] = 3.4e38f; mini[j] = 0; }

#pragma unroll
    for (int kt = 0; kt < 16; ++kt) {
        const int k = kt * 16 + c;                   // this lane's code column
        v8f acc = {};
#pragma unroll
        for (int dstep = 0; dstep < 2; ++dstep) {
            const __bf16* bb = cbw + (((size_t)m * K_CODES + k) * DSUB + dstep * 32 + h * 8);
            v8bf lo = *(const v8bf*)(bb);
            v8bf hi = *(const v8bf*)(bb + 16);
            v16bf bfrag;
#pragma unroll
            for (int i = 0; i < 8; ++i) { bfrag[i] = lo[i]; bfrag[i + 8] = hi[i]; }
            acc = __builtin_amdgcn_wmma_f32_16x16x32_bf16(
                      false, afrag[dstep], false, bfrag,
                      (short)0, acc, false, false);
        }
        const float c2k = c2w[m * K_CODES + k];
#pragma unroll
        for (int j = 0; j < 8; ++j) {
            // x^2 term is constant per row -> irrelevant for argmin
            float val = __builtin_fmaf(-2.0f, acc[j], c2k);
            if (val < minv[j]) { minv[j] = val; mini[j] = k; }
        }
    }

    // min-reduce across the 16 lanes that share each row (wave32 butterfly)
#pragma unroll
    for (int off = 1; off < 16; off <<= 1) {
#pragma unroll
        for (int j = 0; j < 8; ++j) {
            float ov = __shfl_xor(minv[j], off, 32);
            int   oi = __shfl_xor(mini[j], off, 32);
            if (ov < minv[j] || (ov == minv[j] && oi < mini[j])) {
                minv[j] = ov; mini[j] = oi;
            }
        }
    }

    if (c == 0) {                                    // lanes 0 and 16
#pragma unroll
        for (int j = 0; j < 8; ++j) {
            int row = j + 8 * h;
            codes_out[(size_t)(n0 + row) * M_SUB + m] = mini[j];
        }
    }
}

// ---------------------------------------------------------------------------
// Kernel 2: gather chosen codewords (exact f32) -> x_recon and side_output
// 16 lanes per (n,m) pair, float4 (b128) loads/stores.
// ---------------------------------------------------------------------------
__global__ __launch_bounds__(256) void pq_gather_kernel(
    const float4* __restrict__ codebook4, const int* __restrict__ codes,
    float4* __restrict__ xr, float4* __restrict__ side)
{
    int gid  = blockIdx.x * 256 + threadIdx.x;
    int sub  = gid & 15;           // which float4 of the 64-float codeword
    int pair = gid >> 4;
    int m    = pair & 7;
    int n    = pair >> 3;
    int code = codes[n * M_SUB + m];
    float4 v = codebook4[(m * K_CODES + code) * (DSUB / 4) + sub];
    xr  [n * (D_TOTAL / 4) + m * (DSUB / 4) + sub]   = v;
    side[((size_t)m * N_TOTAL + n) * (DSUB / 4) + sub] = v;
}

// ---------------------------------------------------------------------------
extern "C" void kernel_launch(void* const* d_in, const int* in_sizes, int n_in,
                              void* d_out, int out_size, void* d_ws, size_t ws_size,
                              hipStream_t stream) {
    const float* x  = (const float*)d_in[0];           // (N, D) f32
    const float* cb = (const float*)d_in[1];           // (M, K, DSUB) f32

    float* out   = (float*)d_out;
    float* xr    = out;                                            // N*D f32
    int*   codes = (int*)(out + (size_t)N_TOTAL * D_TOTAL);        // N*M int32
    float* side  = out + (size_t)N_TOTAL * D_TOTAL + (size_t)N_TOTAL * M_SUB;

    __bf16* cbw = (__bf16*)d_ws;                                   // 256 KB
    float*  c2w = (float*)((char*)d_ws +
                           (size_t)M_SUB * K_CODES * DSUB * sizeof(__bf16)); // 8 KB

    pq_prep_kernel  <<<(M_SUB * K_CODES + 255) / 256, 256, 0, stream>>>(cb, cbw, c2w);
    pq_codes_kernel <<<N_TOTAL / 16,                   256, 0, stream>>>(x, cbw, c2w, codes);
    pq_gather_kernel<<<(N_TOTAL * M_SUB * 16) / 256,   256, 0, stream>>>(
        (const float4*)cb, codes, (float4*)xr, (float4*)side);
}
